// Model_17085379903578
// MI455X (gfx1250) — matrix operand order repaired and checked by reading, not compiled, not run on hardware
//
#include <hip/hip_runtime.h>
#include <hip/hip_bf16.h>
#include <math.h>
#include <stdint.h>

// ---------------------------------------------------------------------------
// Types for CDNA5 WMMA (wave32): D(16x16,f32) = A(16x32,f16) x B(32x16,f16) + C
// ---------------------------------------------------------------------------
typedef __attribute__((ext_vector_type(16))) _Float16 v16h;
typedef __attribute__((ext_vector_type(8)))  float    v8f;

#define DM 512   // D_MODEL
#define NH 8     // N_HEADS
#define DH 64    // head dim
#define BB 8     // batch

// A-fragment: p points at tile origin (row-major, leading dim ld), 16(M)x32(K)
__device__ inline v16h load_frag_a(const _Float16* p, int ld, int lane) {
  int hf = lane >> 4, m = lane & 15;
  v16h a;
#pragma unroll
  for (int v = 0; v < 8; ++v) {
    int kk = hf * 16 + 2 * v;
    a[2 * v]     = p[m * ld + kk];
    a[2 * v + 1] = p[m * ld + kk + 1];
  }
  return a;
}

// B-fragment: p points at tile origin of a [N][K] (column-major-B) LDS tile,
// 32(K)x16(N). Lane half selects K block of 16; VGPR v holds K=hf*16+2v,+1.
__device__ inline v16h load_frag_b(const _Float16* p, int ld, int lane) {
  int hf = lane >> 4, n = lane & 15;
  v16h b;
#pragma unroll
  for (int v = 0; v < 8; ++v) {
    int kk = hf * 16 + 2 * v;
    b[2 * v]     = p[n * ld + kk];
    b[2 * v + 1] = p[n * ld + kk + 1];
  }
  return b;
}

__device__ inline v8f wmma_f16(v16h a, v16h b, v8f c) {
  return __builtin_amdgcn_wmma_f32_16x16x32_f16(false, a, false, b, (short)0, c,
                                                false, false);
}

// Async global->LDS copy of 16 bytes (ASYNCcnt-tracked). GV addressing mode.
__device__ inline void async_copy_b128(void* lds_dst, const void* gsrc) {
  unsigned lds = (unsigned)(uintptr_t)lds_dst;          // low 32b = LDS offset
  unsigned long long ga = (unsigned long long)(uintptr_t)gsrc;
  asm volatile("global_load_async_to_lds_b128 %0, %1, off"
               :: "v"(lds), "v"(ga) : "memory");
}
__device__ inline void wait_async0() {
  asm volatile("s_wait_asynccnt 0x0" ::: "memory");
}

// ---------------------------------------------------------------------------
// Generic GEMM: C[M,N] = act(A[M,K] @ W[K,N] + bias [+ res]).  M,N mult of 64,
// K mult of 32. act: 0=none, 1=exact GELU. 256 threads, 8 waves, 64x64 tile.
// Double-buffered async global->LDS staging (f32), LDS-side convert to f16
// operand tiles, next tile's async copies issued before current WMMAs.
// ---------------------------------------------------------------------------
__global__ void __launch_bounds__(256) gemm_wmma_kernel(
    const float* __restrict__ A, const float* __restrict__ W,
    const float* __restrict__ bias, const float* __restrict__ res,
    float* __restrict__ C, int M, int N, int K, int act) {
  __shared__ __align__(16) float Afs[2][64 * 32];  // staged A tile [m][k]
  __shared__ __align__(16) float Bfs[2][32 * 64];  // staged W tile [k][n]
  __shared__ __align__(16) _Float16 As[64][32];    // [m][k] f16
  __shared__ __align__(16) _Float16 Bs[64][32];    // [n][k] f16
  int m0 = blockIdx.y * 64, n0 = blockIdx.x * 64;
  int tid = threadIdx.x, lane = tid & 31, wave = tid >> 5;
  int wr = wave >> 1, wc = wave & 1;
  int hf = lane >> 4, lm = lane & 15;
  v8f acc0 = {}; v8f acc1 = {};

  auto issue_tile = [&](int kt, int buf) {
    // A tile: 64 rows x 32 floats (contiguous rows) = 512 b128 chunks
#pragma unroll
    for (int e = tid; e < 512; e += 256) {
      int r = e >> 3, c4 = (e & 7) * 4;
      async_copy_b128(&Afs[buf][r * 32 + c4],
                      A + (size_t)(m0 + r) * K + kt + c4);
    }
    // W tile: 32 rows x 64 floats (contiguous rows) = 512 b128 chunks
#pragma unroll
    for (int e = tid; e < 512; e += 256) {
      int r = e >> 4, c4 = (e & 15) * 4;
      async_copy_b128(&Bfs[buf][r * 64 + c4],
                      W + (size_t)(kt + r) * N + n0 + c4);
    }
  };

  issue_tile(0, 0);
  int nk = K >> 5;
  for (int it = 0; it < nk; ++it) {
    int buf = it & 1;
    wait_async0();     // this wave's async writes to LDS complete
    __syncthreads();   // all waves' staged data visible; f16 tiles reusable
    // convert staged f32 -> f16 operand tiles (B transposed to [n][k])
    for (int e = tid; e < 2048; e += 256) {
      As[e >> 5][e & 31] = (_Float16)Afs[buf][e];
      Bs[e & 63][e >> 6] = (_Float16)Bfs[buf][e];
    }
    __syncthreads();
    if (it + 1 < nk) issue_tile((it + 1) * 32, buf ^ 1);  // overlap with WMMA
    v16h a  = load_frag_a(&As[wr * 16][0], 32, lane);
    v16h b0 = load_frag_b(&Bs[(wc * 2 + 0) * 16][0], 32, lane);
    v16h b1 = load_frag_b(&Bs[(wc * 2 + 1) * 16][0], 32, lane);
    acc0 = wmma_f16(a, b0, acc0);
    acc1 = wmma_f16(a, b1, acc1);
  }
#pragma unroll
  for (int t = 0; t < 2; ++t) {
    v8f acc = t ? acc1 : acc0;
    int n = n0 + (wc * 2 + t) * 16 + lm;
#pragma unroll
    for (int v = 0; v < 8; ++v) {
      int m = m0 + wr * 16 + hf * 8 + v;
      float val = acc[v] + bias[n];
      if (res) val += res[(size_t)m * N + n];
      if (act == 1) val = 0.5f * val * (1.f + erff(val * 0.70710678118f));
      C[(size_t)m * N + n] = val;
    }
  }
}

// ---------------------------------------------------------------------------
// Embedding: circular conv1d(k=3, Cin=7) + sinusoidal pos + time linear.
// ---------------------------------------------------------------------------
__global__ void __launch_bounds__(256) embed_kernel(
    const float* __restrict__ x, const float* __restrict__ xm,
    const float* __restrict__ convw, const float* __restrict__ timew,
    float* __restrict__ out, int L) {
  int idx = blockIdx.x * 256 + threadIdx.x;
  int tot = BB * L * DM;
  if (idx >= tot) return;
  int d = idx & (DM - 1);
  int bl = idx >> 9;
  int l = bl % L, b = bl / L;
  float acc = 0.f;
  for (int t = 0; t < 3; ++t) {
    int ls = (l + t - 1 + L) % L;
    const float* xr = x + ((size_t)(b * L + ls)) * 7;
    const float* wr = convw + (size_t)t * 7 * DM + d;
    for (int c = 0; c < 7; ++c) acc += xr[c] * wr[c * DM];
  }
  int i2 = d >> 1;
  float freq = expf(-(float)(2 * i2) * (logf(10000.f) / (float)DM));
  float ang = (float)l * freq;
  acc += (d & 1) ? cosf(ang) : sinf(ang);
  const float* mr = xm + ((size_t)(b * L + l)) * 4;
  for (int m = 0; m < 4; ++m) acc += mr[m] * timew[m * DM + d];
  out[((size_t)(b * L + l)) * DM + d] = acc;
}

// ---------------------------------------------------------------------------
// LayerNorm: one wave32 per row of 512; 8 rows per block.
// ---------------------------------------------------------------------------
__global__ void __launch_bounds__(256) layernorm_kernel(
    const float* __restrict__ x, const float* __restrict__ g,
    const float* __restrict__ bta, float* __restrict__ y, int M) {
  int row = blockIdx.x * 8 + (threadIdx.x >> 5);
  int lane = threadIdx.x & 31;
  if (row >= M) return;
  const float* xr = x + (size_t)row * DM;
  float vals[16];
  float s = 0.f;
#pragma unroll
  for (int j = 0; j < 16; ++j) { vals[j] = xr[lane + 32 * j]; s += vals[j]; }
#pragma unroll
  for (int o = 16; o > 0; o >>= 1) s += __shfl_xor(s, o, 32);
  float mu = s * (1.f / DM);
  float v2 = 0.f;
#pragma unroll
  for (int j = 0; j < 16; ++j) { float d = vals[j] - mu; v2 += d * d; }
#pragma unroll
  for (int o = 16; o > 0; o >>= 1) v2 += __shfl_xor(v2, o, 32);
  float rinv = rsqrtf(v2 * (1.f / DM) + 1e-5f);
  float* yr = y + (size_t)row * DM;
#pragma unroll
  for (int j = 0; j < 16; ++j) {
    int c = lane + 32 * j;
    yr[c] = (vals[j] - mu) * rinv * g[c] + bta[c];
  }
}

// ---------------------------------------------------------------------------
// LSH buckets: bucket = argmax([x@rot, -x@rot]) over 32, per (b,h,l).
// ---------------------------------------------------------------------------
__global__ void __launch_bounds__(256) buckets_kernel(
    const float* __restrict__ xb, const float* __restrict__ rot,
    int* __restrict__ bucket, int L) {
  int idx = blockIdx.x * 256 + threadIdx.x;
  int tot = BB * NH * L;
  if (idx >= tot) return;
  int l = idx % L;
  int bh = idx / L;
  int h = bh & (NH - 1), b = bh >> 3;
  const float* xr = xb + ((size_t)(b * L + l)) * DM + h * DH;
  float r[16];
#pragma unroll
  for (int n = 0; n < 16; ++n) r[n] = 0.f;
  for (int d = 0; d < DH; ++d) {
    float xv = xr[d];
    const float* rr = rot + d * 16;
#pragma unroll
    for (int n = 0; n < 16; ++n) r[n] += xv * rr[n];
  }
  float best = r[0]; int bi = 0;
#pragma unroll
  for (int n = 1; n < 16; ++n) if (r[n] > best) { best = r[n]; bi = n; }
#pragma unroll
  for (int n = 0; n < 16; ++n) { float v = -r[n]; if (v > best) { best = v; bi = 16 + n; } }
  bucket[(size_t)bh * L + l] = bi;
}

// ---------------------------------------------------------------------------
// Stable counting sort of buckets (values 0..31) per (b,h); 32 threads/block.
// Emulates stable jnp.argsort.
// ---------------------------------------------------------------------------
__global__ void __launch_bounds__(32) lsh_sort_kernel(
    const int* __restrict__ bq, int* __restrict__ perm, int L) {
  __shared__ int cnt[32];
  __shared__ int offs[33];
  int t = threadIdx.x;
  int base = blockIdx.x * L;
  int c = 0;
  for (int l = 0; l < L; ++l) c += (bq[base + l] == t);
  cnt[t] = c;
  __syncthreads();
  if (t == 0) {
    offs[0] = 0;
    for (int i = 0; i < 32; ++i) offs[i + 1] = offs[i] + cnt[i];
  }
  __syncthreads();
  int o = offs[t];
  for (int l = 0; l < L; ++l)
    if (bq[base + l] == t) perm[base + (o++)] = l;
}

// ---------------------------------------------------------------------------
// Gather sorted q/k/v ([B,H,L,64]) and sorted bucket ids.
// ---------------------------------------------------------------------------
__global__ void __launch_bounds__(256) gather_kernel(
    const float* __restrict__ qb, const float* __restrict__ kb,
    const float* __restrict__ vb, const int* __restrict__ pq,
    const int* __restrict__ pk, const int* __restrict__ bq,
    const int* __restrict__ bk, float* __restrict__ qs, float* __restrict__ ks,
    float* __restrict__ vs, int* __restrict__ bqs, int* __restrict__ bks, int L) {
  int idx = blockIdx.x * 256 + threadIdx.x;
  int tot = BB * NH * L;
  if (idx >= tot) return;
  int pos = idx % L;
  int bh = idx / L;
  int h = bh & (NH - 1), b = bh >> 3;
  int lq = pq[(size_t)bh * L + pos];
  int lk = pk[(size_t)bh * L + pos];
  bqs[(size_t)bh * L + pos] = bq[(size_t)bh * L + lq];
  bks[(size_t)bh * L + pos] = bk[(size_t)bh * L + lk];
  size_t so = ((size_t)bh * L + pos) * DH;
  size_t qsrc = ((size_t)(b * L + lq)) * DM + h * DH;
  size_t ksrc = ((size_t)(b * L + lk)) * DM + h * DH;
  for (int d = 0; d < DH; ++d) {
    qs[so + d] = qb[qsrc + d];
    ks[so + d] = kb[ksrc + d];
    vs[so + d] = vb[ksrc + d];
  }
}

// ---------------------------------------------------------------------------
// LSH chunked attention: per (b,h,chunk of 64). S=Q@K^T (WMMA), bucket mask,
// softmax, O=P@V (WMMA), scatter through permutation into [B*L,512] layout.
// ---------------------------------------------------------------------------
__global__ void __launch_bounds__(256) lsh_chunk_attn_kernel(
    const float* __restrict__ qs, const float* __restrict__ ks,
    const float* __restrict__ vs, const int* __restrict__ bqs,
    const int* __restrict__ bks, const int* __restrict__ pq,
    float* __restrict__ out, int L) {
  __shared__ _Float16 Qh[64][64];  // [i][d]
  __shared__ _Float16 Kh[64][64];  // [j][d]  (B-layout for S)
  __shared__ _Float16 Vt[64][64];  // [d][j]  (B-layout for O)
  __shared__ _Float16 Ph[64][64];  // probs
  __shared__ float    Sf[64][64];
  __shared__ int bqsh[64], bksh[64];
  int c = blockIdx.x, h = blockIdx.y, b = blockIdx.z;
  int tid = threadIdx.x, lane = tid & 31, wave = tid >> 5;
  int hf = lane >> 4, lm = lane & 15;
  int wr = wave >> 1, wc = wave & 1;
  int bh = b * NH + h;
  size_t base = (((size_t)bh) * L + c * 64) * DH;
  for (int e = tid; e < 4096; e += 256) {
    int i = e >> 6, d = e & 63;
    Qh[i][d] = (_Float16)qs[base + e];
    Kh[i][d] = (_Float16)ks[base + e];
    Vt[d][i] = (_Float16)vs[base + e];
  }
  if (tid < 64) {
    bqsh[tid] = bqs[(size_t)bh * L + c * 64 + tid];
    bksh[tid] = bks[(size_t)bh * L + c * 64 + tid];
  }
  __syncthreads();
  // --- scores ---
#pragma unroll
  for (int t = 0; t < 2; ++t) {
    int tn = wc * 2 + t;
    v8f acc = {};
#pragma unroll
    for (int k0 = 0; k0 < 64; k0 += 32) {
      v16h a = load_frag_a(&Qh[wr * 16][k0], 64, lane);
      v16h bb = load_frag_b(&Kh[tn * 16][k0], 64, lane);
      acc = wmma_f16(a, bb, acc);
    }
#pragma unroll
    for (int v = 0; v < 8; ++v) {
      int i = wr * 16 + hf * 8 + v;
      int j = tn * 16 + lm;
      Sf[i][j] = (bqsh[i] == bksh[j]) ? acc[v] * 0.125f : -1e9f;
    }
  }
  __syncthreads();
  // --- softmax rows ---
  if (tid < 64) {
    int r = tid;
    float mx = -3.0e38f;
    for (int j = 0; j < 64; ++j) mx = fmaxf(mx, Sf[r][j]);
    float sm = 0.f;
    for (int j = 0; j < 64; ++j) { float e = expf(Sf[r][j] - mx); Sf[r][j] = e; sm += e; }
    float inv = 1.f / sm;
    for (int j = 0; j < 64; ++j) Ph[r][j] = (_Float16)(Sf[r][j] * inv);
  }
  __syncthreads();
  // --- O = P @ V, scatter back ---
#pragma unroll
  for (int t = 0; t < 2; ++t) {
    int tn = wc * 2 + t;
    v8f acc = {};
#pragma unroll
    for (int k0 = 0; k0 < 64; k0 += 32) {
      v16h a = load_frag_a(&Ph[wr * 16][k0], 64, lane);
      v16h bb = load_frag_b(&Vt[tn * 16][k0], 64, lane);
      acc = wmma_f16(a, bb, acc);
    }
#pragma unroll
    for (int v = 0; v < 8; ++v) {
      int i = wr * 16 + hf * 8 + v;
      int pos = c * 64 + i;
      int l = pq[(size_t)bh * L + pos];
      int d = tn * 16 + lm;
      out[((size_t)(b * L + l)) * DM + h * DH + d] = acc[v];
    }
  }
}

// ---------------------------------------------------------------------------
// Full cross-attention (decoder): streaming softmax over Lk, WMMA S and O.
// q: [B*Lq,512] head-cols; k,v: [B*Lk,512]; out: [B*Lq,512].
// exp without max-subtraction (scores are O(0.2), safe).
// ---------------------------------------------------------------------------
__global__ void __launch_bounds__(256) cross_attn_kernel(
    const float* __restrict__ qb, const float* __restrict__ kb,
    const float* __restrict__ vb, float* __restrict__ out, int Lq, int Lk) {
  __shared__ _Float16 Qh[64][64];
  __shared__ _Float16 Kh[64][64];
  __shared__ _Float16 Vt[64][64];
  __shared__ _Float16 Ph[64][64];
  __shared__ float    Sf[64][64];
  __shared__ float    den[64];
  int qt = blockIdx.x, h = blockIdx.y, b = blockIdx.z;
  int tid = threadIdx.x, lane = tid & 31, wave = tid >> 5;
  int hf = lane >> 4, lm = lane & 15;
  int wr = wave >> 1, wc = wave & 1;
  int q0 = qt * 64;
  for (int e = tid; e < 4096; e += 256) {
    int i = e >> 6, d = e & 63;
    Qh[i][d] = (_Float16)qb[((size_t)(b * Lq + q0 + i)) * DM + h * DH + d];
  }
  if (tid < 64) den[tid] = 0.f;
  v8f o0 = {}; v8f o1 = {};
  for (int kc = 0; kc < Lk; kc += 64) {
    __syncthreads();  // protects Qh (first iter) and Kh/Vt/Ph from consumers
    for (int e = tid; e < 4096; e += 256) {
      int j = e >> 6, d = e & 63;
      size_t src = ((size_t)(b * Lk + kc + j)) * DM + h * DH + d;
      Kh[j][d] = (_Float16)kb[src];
      Vt[d][j] = (_Float16)vb[src];
    }
    __syncthreads();
#pragma unroll
    for (int t = 0; t < 2; ++t) {
      int tn = wc * 2 + t;
      v8f acc = {};
#pragma unroll
      for (int k0 = 0; k0 < 64; k0 += 32) {
        v16h a = load_frag_a(&Qh[wr * 16][k0], 64, lane);
        v16h bb = load_frag_b(&Kh[tn * 16][k0], 64, lane);
        acc = wmma_f16(a, bb, acc);
      }
#pragma unroll
      for (int v = 0; v < 8; ++v) {
        int i = wr * 16 + hf * 8 + v;
        int j = tn * 16 + lm;
        Sf[i][j] = expf(acc[v] * 0.125f);
      }
    }
    __syncthreads();
    if (tid < 64) {
      int r = tid;
      float sm = 0.f;
      for (int j = 0; j < 64; ++j) { float e = Sf[r][j]; Ph[r][j] = (_Float16)e; sm += e; }
      den[r] += sm;
    }
    __syncthreads();
#pragma unroll
    for (int t = 0; t < 2; ++t) {
      int tn = wc * 2 + t;
      v8f acc = t ? o1 : o0;
#pragma unroll
      for (int k0 = 0; k0 < 64; k0 += 32) {
        v16h a = load_frag_a(&Ph[wr * 16][k0], 64, lane);
        v16h bb = load_frag_b(&Vt[tn * 16][k0], 64, lane);
        acc = wmma_f16(a, bb, acc);
      }
      if (t) o1 = acc; else o0 = acc;
    }
  }
  __syncthreads();
#pragma unroll
  for (int t = 0; t < 2; ++t) {
    int tn = wc * 2 + t;
    v8f acc = t ? o1 : o0;
#pragma unroll
    for (int v = 0; v < 8; ++v) {
      int i = wr * 16 + hf * 8 + v;
      int d = tn * 16 + lm;
      out[((size_t)(b * Lq + q0 + i)) * DM + h * DH + d] = acc[v] / den[i];
    }
  }
}

// ---------------------------------------------------------------------------
// Final projection: out[b,t,c] = dec[b, 256+t] @ proj_w[:,c] + proj_b[c]
// ---------------------------------------------------------------------------
__global__ void __launch_bounds__(256) proj_kernel(
    const float* __restrict__ xd, const float* __restrict__ pw,
    const float* __restrict__ pb, float* __restrict__ out) {
  int idx = blockIdx.x * 256 + threadIdx.x;
  if (idx >= 8 * 256 * 7) return;
  int c = idx % 7;
  int t = (idx / 7) % 256;
  int b = idx / (7 * 256);
  const float* xr = xd + ((size_t)(b * 512 + 256 + t)) * DM;
  float acc = pb[c];
  for (int d = 0; d < DM; ++d) acc += xr[d] * pw[d * 7 + c];
  out[idx] = acc;
}

// ---------------------------------------------------------------------------
// Host launch. Param leaf order = JAX pytree flatten (dicts sorted by key):
//  0:x_enc 1:x_mark_enc 2:x_dec 3:x_mark_dec
//  4..29 dec[0]: cross{bk,bo,bq,bv,wk,wo,wq,wv}=4..11, ff1_b=12,ff1_w=13,
//    ff2_b=14,ff2_w=15, ln1_b=16,ln1_g=17,ln2_b=18,ln2_g=19,ln3_b=20,ln3_g=21,
//    self{bk,bo,bq,bv,wk,wo,wq,wv}=22..29
//  30:dec_norm_b 31:dec_norm_g 32:emb_dec.conv_w 33:emb_dec.time_w
//  34:emb_enc.conv_w 35:emb_enc.time_w
//  36..51 enc[0]: attn{bk,bo,bq,bv,wk,wo,wq,wv}=+0..7, ff1_b=+8,ff1_w=+9,
//    ff2_b=+10,ff2_w=+11, ln1_b=+12,ln1_g=+13,ln2_b=+14,ln2_g=+15
//  52..67 enc[1]  68:enc_norm_b 69:enc_norm_g 70:proj_b 71:proj_w 72:rot
// ---------------------------------------------------------------------------
extern "C" void kernel_launch(void* const* d_in, const int* in_sizes, int n_in,
                              void* d_out, int out_size, void* d_ws,
                              size_t ws_size, hipStream_t stream) {
  (void)in_sizes; (void)n_in; (void)out_size; (void)ws_size;
  auto F = [&](int i) { return (const float*)d_in[i]; };

  float* w = (float*)d_ws;
  size_t o = 0;
  auto alloc = [&](size_t n) { float* p = w + o; o += n; return p; };
  float* XE = alloc(4194304);   // [8*1024,512]
  float* X2 = alloc(4194304);
  float* Qb = alloc(4194304);
  float* Kb = alloc(4194304);
  float* Vb = alloc(4194304);
  float* H1 = alloc(16777216);  // [8*1024,2048]
  float* QS = alloc(4194304);
  float* KS = alloc(4194304);
  float* VS = alloc(4194304);
  float* AT = alloc(4194304);
  float* XD = alloc(2097152);   // [8*512,512]
  int* IB = (int*)(w + o);
  int* BQ = IB;            int* BK = IB + 65536;
  int* PQ = IB + 131072;   int* PK = IB + 196608;
  int* BQS = IB + 262144;  int* BKS = IB + 327680;
  const float* rot = F(72);

  auto gemm = [&](const float* A, const float* Wt, const float* bias,
                  const float* res, float* C, int M, int N, int K, int act) {
    dim3 g(N / 64, M / 64);
    gemm_wmma_kernel<<<g, 256, 0, stream>>>(A, Wt, bias, res, C, M, N, K, act);
  };
  auto ln = [&](const float* xx, const float* gg, const float* bb, float* yy,
                int M) {
    layernorm_kernel<<<M / 8, 256, 0, stream>>>(xx, gg, bb, yy, M);
  };
  auto lsh_attn = [&](const float* X, int L, const float* wq, const float* bq,
                      const float* wk, const float* bk, const float* wv,
                      const float* bv) {
    int M = BB * L;
    gemm(X, wq, bq, nullptr, Qb, M, 512, 512, 0);
    gemm(X, wk, bk, nullptr, Kb, M, 512, 512, 0);
    gemm(X, wv, bv, nullptr, Vb, M, 512, 512, 0);
    int tot = BB * NH * L;
    buckets_kernel<<<(tot + 255) / 256, 256, 0, stream>>>(Qb, rot, BQ, L);
    buckets_kernel<<<(tot + 255) / 256, 256, 0, stream>>>(Kb, rot, BK, L);
    lsh_sort_kernel<<<BB * NH, 32, 0, stream>>>(BQ, PQ, L);
    lsh_sort_kernel<<<BB * NH, 32, 0, stream>>>(BK, PK, L);
    gather_kernel<<<(tot + 255) / 256, 256, 0, stream>>>(
        Qb, Kb, Vb, PQ, PK, BQ, BK, QS, KS, VS, BQS, BKS, L);
    lsh_chunk_attn_kernel<<<dim3(L / 64, NH, BB), 256, 0, stream>>>(
        QS, KS, VS, BQS, BKS, PQ, AT, L);
  };

  // ===== Encoder =====
  embed_kernel<<<(8 * 1024 * 512 + 255) / 256, 256, 0, stream>>>(
      F(0), F(1), F(34), F(35), XE, 1024);
  for (int layer = 0; layer < 2; ++layer) {
    int base = 36 + layer * 16;
    lsh_attn(XE, 1024, F(base + 6), F(base + 2), F(base + 4), F(base + 0),
             F(base + 7), F(base + 3));                       // wq,bq,wk,bk,wv,bv
    gemm(AT, F(base + 5), F(base + 1), XE, X2, 8192, 512, 512, 0);  // wo,bo,+res
    ln(X2, F(base + 13), F(base + 12), XE, 8192);             // ln1
    gemm(XE, F(base + 9), F(base + 8), nullptr, H1, 8192, 2048, 512, 1);  // ff1+gelu
    gemm(H1, F(base + 11), F(base + 10), XE, X2, 8192, 512, 2048, 0);     // ff2+res
    ln(X2, F(base + 15), F(base + 14), XE, 8192);             // ln2
  }
  ln(XE, F(69), F(68), XE, 8192);  // enc_norm

  // ===== Decoder =====
  embed_kernel<<<(8 * 512 * 512 + 255) / 256, 256, 0, stream>>>(
      F(2), F(3), F(32), F(33), XD, 512);
  // self LSH attention
  lsh_attn(XD, 512, F(28), F(24), F(26), F(22), F(29), F(25));
  gemm(AT, F(27), F(23), XD, X2, 4096, 512, 512, 0);
  ln(X2, F(17), F(16), XD, 4096);  // ln1
  // cross attention
  gemm(XD, F(10), F(6), nullptr, Qb, 4096, 512, 512, 0);   // cross wq,bq
  gemm(XE, F(8), F(4), nullptr, Kb, 8192, 512, 512, 0);    // cross wk,bk
  gemm(XE, F(11), F(7), nullptr, Vb, 8192, 512, 512, 0);   // cross wv,bv
  cross_attn_kernel<<<dim3(512 / 64, NH, BB), 256, 0, stream>>>(Qb, Kb, Vb, AT,
                                                                512, 1024);
  gemm(AT, F(9), F(5), XD, X2, 4096, 512, 512, 0);         // cross wo,bo,+res
  ln(X2, F(19), F(18), XD, 4096);  // ln2
  // FFN
  gemm(XD, F(13), F(12), nullptr, H1, 4096, 2048, 512, 1);
  gemm(H1, F(15), F(14), XD, X2, 4096, 512, 2048, 0);
  ln(X2, F(21), F(20), XD, 4096);  // ln3
  ln(XD, F(31), F(30), XD, 4096);  // dec_norm
  proj_kernel<<<(8 * 256 * 7 + 255) / 256, 256, 0, stream>>>(
      XD, F(71), F(70), (float*)d_out);
}
